// GNN_encoder_2757369004123
// MI455X (gfx1250) — matrix operand order repaired and checked by reading, not compiled, not run on hardware
//
#include <hip/hip_runtime.h>
#include <hip/hip_bf16.h>

// ---------------------------------------------------------------------------
// Problem constants (match reference)
// ---------------------------------------------------------------------------
#define NN   16384
#define EE   131072
#define FIN  16
#define SS   32
#define FOUT 64
#define HID  32
#define FCN  32

#define KSLOT   552   // padded row stride (halves) for packed W [64][KSLOT]
#define KCHUNKS 17    // 17*32 = 544 >= 512 (W2) + 16 (b2 rows)

typedef __attribute__((ext_vector_type(16))) _Float16 v16h;
typedef __attribute__((ext_vector_type(8)))  _Float16 v8h;
typedef __attribute__((ext_vector_type(8)))  float    v8f;

__device__ __forceinline__ void atomAddF(float* p, float v) {
    unsafeAtomicAdd(p, v);   // hardware global_atomic_add_f32
}

__device__ __forceinline__ v16h loadB(const _Float16* p) {
    const v8h* bp = (const v8h*)p;
    return __builtin_shufflevector(bp[0], bp[2],
        0, 1, 2, 3, 4, 5, 6, 7, 8, 9, 10, 11, 12, 13, 14, 15);
}

// ---------------------------------------------------------------------------
// Pack W2 [32,1024] + b2 [1024] into f16, o-major: wcomb[o][k], k = j*16+i
// rows 512..527 = b2 (paired with x_src in the P matrix), 528..551 = 0 pad.
// ---------------------------------------------------------------------------
__global__ void build_wcomb(const float* __restrict__ W2,
                            const float* __restrict__ b2,
                            _Float16* __restrict__ wcomb) {
    int idx = blockIdx.x * 256 + threadIdx.x;
    if (idx >= 64 * KSLOT) return;
    int o = idx / KSLOT;
    int k = idx % KSLOT;
    float v = 0.0f;
    if (k < 512) {
        int j = k >> 4, i = k & 15;
        v = W2[(size_t)j * 1024 + i * 64 + o];
    } else if (k < 528) {
        int i = k - 512;
        v = b2[i * 64 + o];
    }
    wcomb[idx] = (_Float16)v;
}

// ---------------------------------------------------------------------------
// h = relu(e @ W1 + b1) -> f16 [E,32].  LDS-tiled, 32 edges per block.
// ---------------------------------------------------------------------------
__global__ __launch_bounds__(256)
void h_mlp(const float* __restrict__ eFeat, const float* __restrict__ W1,
           const float* __restrict__ b1, _Float16* __restrict__ h16) {
    __shared__ float eL[32][33];
    __shared__ float wL[32][33];
    int base = blockIdx.x * 32;
    for (int idx = threadIdx.x; idx < 1024; idx += 256) {
        int r = idx >> 5, c = idx & 31;
        wL[r][c] = W1[idx];
        eL[r][c] = eFeat[(size_t)(base + r) * SS + c];
    }
    __syncthreads();
    for (int idx = threadIdx.x; idx < 1024; idx += 256) {
        int m = idx >> 5, j = idx & 31;
        float acc = b1[j];
#pragma unroll
        for (int k = 0; k < 32; ++k) acc += eL[m][k] * wL[k][j];
        h16[(size_t)(base + m) * HID + j] = (_Float16)fmaxf(acc, 0.0f);
    }
}

// ---------------------------------------------------------------------------
// In-degree counts (for scatter-mean).
// ---------------------------------------------------------------------------
__global__ void count_deg(const int* __restrict__ dst, int* __restrict__ cnt) {
    int e = blockIdx.x * 256 + threadIdx.x;
    if (e < EE) atomicAdd(&cnt[dst[e]], 1);
}

// ---------------------------------------------------------------------------
// Main WMMA kernel: msg = P @ Wcomb,  P[e, j*16+i] = h[e,j]*x[src[e],i]
// (rows 512..527 of P are x_src itself -> adds the b2 term).
// Each wave processes TWO 16-edge tiles so every B fragment loaded from LDS
// feeds two WMMAs (halves LDS bandwidth per WMMA and doubles independent
// matrix ops to hide ds latency + WMMA->VALU hazards). 8 waves/block share
// the 69KB packed-W in dynamic LDS.  Scatter-add results into agg[dst].
// ---------------------------------------------------------------------------
__global__ __launch_bounds__(256)
void ecc_edge_wmma(const float* __restrict__ x,
                   const int* __restrict__ src,
                   const int* __restrict__ dst,
                   const _Float16* __restrict__ h16,
                   const _Float16* __restrict__ wcomb,
                   float* __restrict__ agg) {
    extern __shared__ _Float16 wlds[];   // 64*KSLOT halves
    {
        const uint32_t* s = (const uint32_t*)wcomb;
        uint32_t* d = (uint32_t*)wlds;
        const int nw = 64 * KSLOT / 2;   // 17664 dwords
        for (int idx = threadIdx.x; idx < nw; idx += 256) d[idx] = s[idx];
    }
    __syncthreads();

    const int wave = threadIdx.x >> 5;
    const int lane = threadIdx.x & 31;
    const int hh   = lane >> 4;          // lane half (0/1)
    const int col  = lane & 15;
    const int tileBase = (blockIdx.x * 8 + wave) * 32;   // 32 edges per wave
    const int e0 = tileBase + col;        // A-row, tile 0
    const int e1 = tileBase + 16 + col;   // A-row, tile 1

    // h rows in registers (+ [32]=1 for the b2 block, [33]=0 pad)
    _Float16 hreg0[34], hreg1[34];
    {
        const v8h* hp0 = (const v8h*)(h16 + (size_t)e0 * HID);
        const v8h* hp1 = (const v8h*)(h16 + (size_t)e1 * HID);
        v8h a0 = hp0[0], a1 = hp0[1], a2 = hp0[2], a3 = hp0[3];
        v8h c0 = hp1[0], c1 = hp1[1], c2 = hp1[2], c3 = hp1[3];
#pragma unroll
        for (int q = 0; q < 8; ++q) {
            hreg0[q] = a0[q]; hreg0[8 + q] = a1[q];
            hreg0[16 + q] = a2[q]; hreg0[24 + q] = a3[q];
            hreg1[q] = c0[q]; hreg1[8 + q] = c1[q];
            hreg1[16 + q] = c2[q]; hreg1[24 + q] = c3[q];
        }
        hreg0[32] = (_Float16)1.0f; hreg0[33] = (_Float16)0.0f;
        hreg1[32] = (_Float16)1.0f; hreg1[33] = (_Float16)0.0f;
    }
    // the 8 gathered x values each lane needs per tile: i = 8*hh + q
    _Float16 xh0[8], xh1[8];
    {
        int s0 = src[e0];
        int s1 = src[e1];
        const float4* xp0 = (const float4*)(x + (size_t)s0 * FIN + hh * 8);
        const float4* xp1 = (const float4*)(x + (size_t)s1 * FIN + hh * 8);
        float4 xa = xp0[0], xb = xp0[1];
        float4 xc = xp1[0], xd = xp1[1];
        xh0[0] = (_Float16)xa.x; xh0[1] = (_Float16)xa.y;
        xh0[2] = (_Float16)xa.z; xh0[3] = (_Float16)xa.w;
        xh0[4] = (_Float16)xb.x; xh0[5] = (_Float16)xb.y;
        xh0[6] = (_Float16)xb.z; xh0[7] = (_Float16)xb.w;
        xh1[0] = (_Float16)xc.x; xh1[1] = (_Float16)xc.y;
        xh1[2] = (_Float16)xc.z; xh1[3] = (_Float16)xc.w;
        xh1[4] = (_Float16)xd.x; xh1[5] = (_Float16)xd.y;
        xh1[6] = (_Float16)xd.z; xh1[7] = (_Float16)xd.w;
    }

    v8f p0_0 = {}, p0_1 = {}, p0_2 = {}, p0_3 = {};   // tile 0 accumulators
    v8f p1_0 = {}, p1_1 = {}, p1_2 = {}, p1_3 = {};   // tile 1 accumulators

#pragma unroll
    for (int kc = 0; kc < KCHUNKS; ++kc) {
        const int kbase = kc * 32 + hh * 8;   // B-frag k base (halves)
        // B fragments for this K chunk (shared by both tiles)
        v16h b0 = loadB(wlds + (size_t)(0 * 16 + col) * KSLOT + kbase);
        v16h b1 = loadB(wlds + (size_t)(1 * 16 + col) * KSLOT + kbase);
        v16h b2 = loadB(wlds + (size_t)(2 * 16 + col) * KSLOT + kbase);
        v16h b3 = loadB(wlds + (size_t)(3 * 16 + col) * KSLOT + kbase);
        // A fragments: a[t] = hreg[2*kc + (t>>3)] * xh[t&7]  (indices are
        // compile-time after unroll; i-half selection folded into xh load)
        v16h a0, a1;
#pragma unroll
        for (int t = 0; t < 16; ++t) {
            a0[t] = hreg0[2 * kc + (t >> 3)] * xh0[t & 7];
            a1[t] = hreg1[2 * kc + (t >> 3)] * xh1[t & 7];
        }
        p0_0 = __builtin_amdgcn_wmma_f32_16x16x32_f16(false, a0, false, b0,
                                                      (short)0, p0_0, false, false);
        p1_0 = __builtin_amdgcn_wmma_f32_16x16x32_f16(false, a1, false, b0,
                                                      (short)0, p1_0, false, false);
        p0_1 = __builtin_amdgcn_wmma_f32_16x16x32_f16(false, a0, false, b1,
                                                      (short)0, p0_1, false, false);
        p1_1 = __builtin_amdgcn_wmma_f32_16x16x32_f16(false, a1, false, b1,
                                                      (short)0, p1_1, false, false);
        p0_2 = __builtin_amdgcn_wmma_f32_16x16x32_f16(false, a0, false, b2,
                                                      (short)0, p0_2, false, false);
        p1_2 = __builtin_amdgcn_wmma_f32_16x16x32_f16(false, a1, false, b2,
                                                      (short)0, p1_2, false, false);
        p0_3 = __builtin_amdgcn_wmma_f32_16x16x32_f16(false, a0, false, b3,
                                                      (short)0, p0_3, false, false);
        p1_3 = __builtin_amdgcn_wmma_f32_16x16x32_f16(false, a1, false, b3,
                                                      (short)0, p1_3, false, false);
    }

    // C/D layout: acc[r] holds m = r + 8*hh, n = col (+16*nt). Scatter-add.
#pragma unroll
    for (int r = 0; r < 8; ++r) {
        int m = r + 8 * hh;
        int d0 = dst[tileBase + m];
        float* base0 = agg + (size_t)d0 * FOUT + col;
        atomAddF(base0 + 0,  p0_0[r]);
        atomAddF(base0 + 16, p0_1[r]);
        atomAddF(base0 + 32, p0_2[r]);
        atomAddF(base0 + 48, p0_3[r]);
        int d1 = dst[tileBase + 16 + m];
        float* base1 = agg + (size_t)d1 * FOUT + col;
        atomAddF(base1 + 0,  p1_0[r]);
        atomAddF(base1 + 16, p1_1[r]);
        atomAddF(base1 + 32, p1_2[r]);
        atomAddF(base1 + 48, p1_3[r]);
    }
}

// ---------------------------------------------------------------------------
// Node update: out = relu(agg/max(cnt,1) + x@root + b_ecc) (kept in LDS),
// xg = out@Wg, att_s = xg.a_self, att_n = xg.a_neigh.  4 nodes / block.
// ---------------------------------------------------------------------------
__global__ __launch_bounds__(256)
void node_update(const float* __restrict__ x, const float* __restrict__ agg,
                 const int* __restrict__ cnt, const float* __restrict__ root,
                 const float* __restrict__ b_ecc, const float* __restrict__ Wg,
                 const float* __restrict__ a_self, const float* __restrict__ a_neigh,
                 float* __restrict__ xg, float* __restrict__ att_s,
                 float* __restrict__ att_n) {
    __shared__ float rootL[16 * 64];
    __shared__ float wgL[64 * 64];
    __shared__ float outL[4][64];
    __shared__ float redL[4][64];
    for (int idx = threadIdx.x; idx < 1024; idx += 256) rootL[idx] = root[idx];
    for (int idx = threadIdx.x; idx < 4096; idx += 256) wgL[idx] = Wg[idx];
    __syncthreads();

    int ln = threadIdx.x >> 6, o = threadIdx.x & 63;
    int n = blockIdx.x * 4 + ln;

    float xr = 0.0f;
#pragma unroll
    for (int i = 0; i < 16; ++i) xr += x[(size_t)n * FIN + i] * rootL[i * 64 + o];
    int c = cnt[n];
    float inv = 1.0f / (float)(c > 1 ? c : 1);
    float v = fmaxf(agg[(size_t)n * FOUT + o] * inv + xr + b_ecc[o], 0.0f);
    outL[ln][o] = v;
    __syncthreads();

    float g = 0.0f;
#pragma unroll
    for (int k = 0; k < 64; ++k) g += outL[ln][k] * wgL[k * 64 + o];
    xg[(size_t)n * FOUT + o] = g;

    redL[ln][o] = g * a_self[o];
    __syncthreads();
    for (int st = 32; st > 0; st >>= 1) {
        if (o < st) redL[ln][o] += redL[ln][o + st];
        __syncthreads();
    }
    if (o == 0) att_s[n] = redL[ln][0];
    __syncthreads();

    redL[ln][o] = g * a_neigh[o];
    __syncthreads();
    for (int st = 32; st > 0; st >>= 1) {
        if (o < st) redL[ln][o] += redL[ln][o + st];
        __syncthreads();
    }
    if (o == 0) att_n[n] = redL[ln][0];
}

// ---------------------------------------------------------------------------
// Softmax denominator over incoming edges (+ self loops). Logits are tiny
// (|lg| << 1) so exp without per-segment max is numerically exact softmax.
// ---------------------------------------------------------------------------
__global__ void edge_den(const int* __restrict__ src, const int* __restrict__ dst,
                         const float* __restrict__ att_s, const float* __restrict__ att_n,
                         float* __restrict__ den) {
    int e = blockIdx.x * 256 + threadIdx.x;
    if (e >= EE + NN) return;
    int s = (e < EE) ? src[e] : (e - EE);
    int d = (e < EE) ? dst[e] : (e - EE);
    float lg = att_s[d] + att_n[s];
    lg = (lg > 0.0f) ? lg : 0.2f * lg;
    atomAddF(&den[d], expf(lg));
}

// ---------------------------------------------------------------------------
// Weighted scatter: ygat[d] += alpha * xg[s]. One (edge, feature) per thread.
// ---------------------------------------------------------------------------
__global__ void edge_gat(const int* __restrict__ src, const int* __restrict__ dst,
                         const float* __restrict__ att_s, const float* __restrict__ att_n,
                         const float* __restrict__ den, const float* __restrict__ xg,
                         float* __restrict__ ygat) {
    int gid = blockIdx.x * 256 + threadIdx.x;
    int e = gid >> 6, o = gid & 63;
    if (e >= EE + NN) return;
    int s = (e < EE) ? src[e] : (e - EE);
    int d = (e < EE) ? dst[e] : (e - EE);
    float lg = att_s[d] + att_n[s];
    lg = (lg > 0.0f) ? lg : 0.2f * lg;
    float alpha = expf(lg) / den[d];
    atomAddF(&ygat[(size_t)d * FOUT + o], alpha * xg[(size_t)s * FOUT + o]);
}

// ---------------------------------------------------------------------------
// Pool stage 1: grid-strided partial sums of relu(ygat + b_gat) per column,
// accumulated into zsum[64] with hardware fp32 atomics (full-HBM-rate read).
// ---------------------------------------------------------------------------
__global__ __launch_bounds__(256)
void pool_partial(const float* __restrict__ ygat, const float* __restrict__ b_gat,
                  float* __restrict__ zsum) {
    __shared__ float red[256];
    int o = threadIdx.x & 63, r = threadIdx.x >> 6;   // 4 rows per block pass
    float bo = b_gat[o];
    float acc = 0.0f;
    for (int n = blockIdx.x * 4 + r; n < NN; n += gridDim.x * 4)
        acc += fmaxf(ygat[(size_t)n * FOUT + o] + bo, 0.0f);
    red[threadIdx.x] = acc;
    __syncthreads();
    if (threadIdx.x < 64) {
        float s = red[o] + red[64 + o] + red[128 + o] + red[192 + o];
        atomAddF(&zsum[o], s);
    }
}

// ---------------------------------------------------------------------------
// Pool stage 2: z = zsum/N, out = relu(z @ Wfc + bfc). 32 threads.
// ---------------------------------------------------------------------------
__global__ void fc_final(const float* __restrict__ zsum, const float* __restrict__ Wfc,
                         const float* __restrict__ bfc, float* __restrict__ out) {
    int c = threadIdx.x;
    if (c >= FCN) return;
    float r = bfc[c];
#pragma unroll
    for (int k = 0; k < 64; ++k) r += (zsum[k] * (1.0f / (float)NN)) * Wfc[k * FCN + c];
    out[c] = fmaxf(r, 0.0f);
}

// ---------------------------------------------------------------------------
extern "C" void kernel_launch(void* const* d_in, const int* in_sizes, int n_in,
                              void* d_out, int out_size, void* d_ws, size_t ws_size,
                              hipStream_t stream) {
    const float* x      = (const float*)d_in[0];
    const float* eFeat  = (const float*)d_in[1];
    const int*   src    = (const int*)d_in[2];
    const int*   dst    = (const int*)d_in[3];
    const float* W1     = (const float*)d_in[4];
    const float* b1     = (const float*)d_in[5];
    const float* W2     = (const float*)d_in[6];
    const float* b2     = (const float*)d_in[7];
    const float* root   = (const float*)d_in[8];
    const float* b_ecc  = (const float*)d_in[9];
    const float* Wg     = (const float*)d_in[10];
    const float* a_self = (const float*)d_in[11];
    const float* a_neigh= (const float*)d_in[12];
    const float* b_gat  = (const float*)d_in[13];
    const float* Wfc    = (const float*)d_in[14];
    const float* bfc    = (const float*)d_in[15];

    char* ws = (char*)d_ws;
    size_t off = 0;
    auto alloc = [&](size_t bytes) -> void* {
        void* p = ws + off;
        off += (bytes + 255) & ~(size_t)255;
        return p;
    };
    _Float16* wcomb = (_Float16*)alloc((size_t)64 * KSLOT * 2);
    _Float16* h16   = (_Float16*)alloc((size_t)EE * HID * 2);
    float*    agg   = (float*)alloc((size_t)NN * FOUT * 4);
    int*      cnt   = (int*)alloc((size_t)NN * 4);
    float*    xg    = (float*)alloc((size_t)NN * FOUT * 4);
    float*    atts  = (float*)alloc((size_t)NN * 4);
    float*    attn  = (float*)alloc((size_t)NN * 4);
    float*    den   = (float*)alloc((size_t)NN * 4);
    float*    ygat  = (float*)alloc((size_t)NN * FOUT * 4);
    float*    zsum  = (float*)alloc((size_t)64 * 4);

    hipMemsetAsync(agg, 0, (size_t)NN * FOUT * 4, stream);
    hipMemsetAsync(cnt, 0, (size_t)NN * 4, stream);
    hipMemsetAsync(den, 0, (size_t)NN * 4, stream);
    hipMemsetAsync(ygat, 0, (size_t)NN * FOUT * 4, stream);
    hipMemsetAsync(zsum, 0, (size_t)64 * 4, stream);

    build_wcomb<<<(64 * KSLOT + 255) / 256, 256, 0, stream>>>(W2, b2, wcomb);
    h_mlp<<<EE / 32, 256, 0, stream>>>(eFeat, W1, b1, h16);
    count_deg<<<EE / 256, 256, 0, stream>>>(dst, cnt);
    ecc_edge_wmma<<<EE / 256, 256, (size_t)64 * KSLOT * 2, stream>>>(
        x, src, dst, h16, wcomb, agg);
    node_update<<<NN / 4, 256, 0, stream>>>(x, agg, cnt, root, b_ecc, Wg,
                                            a_self, a_neigh, xg, atts, attn);
    edge_den<<<(EE + NN + 255) / 256, 256, 0, stream>>>(src, dst, atts, attn, den);
    edge_gat<<<((EE + NN) * 64) / 256, 256, 0, stream>>>(src, dst, atts, attn,
                                                         den, xg, ygat);
    pool_partial<<<256, 256, 0, stream>>>(ygat, b_gat, zsum);
    fc_final<<<1, 64, 0, stream>>>(zsum, Wfc, bfc, (float*)d_out);
}
